// FlaxFalconAttention_33225867002018
// MI455X (gfx1250) — matrix operand order repaired and checked by reading, not compiled, not run on hardware
//
#include <hip/hip_runtime.h>
#include <math.h>

typedef __attribute__((ext_vector_type(16))) __bf16 v16bf;
typedef __attribute__((ext_vector_type(8)))  float  v8f;

#define HIDDEN 4096
#define NH 32
#define HD 128
#define SEQ 2048
#define BATCH 2
#define BS (BATCH * SEQ)

// ---------------------------------------------------------------------------
// WMMA helpers (CDNA5 gfx1250, wave32). Layouts per ISA 7.12.2:
//  A (16x32 bf16): lane row = lane%16; K group0 = half*8+[0..7], group1 = 16+half*8+[0..7]
//  B (32x16 bf16): lane col = lane%16; K group0 = half*8+[0..7], group1 = 16+half*8+[0..7] (staged N-major)
//  C/D (16x16 f32): lane col = lane%16; row = half*8 + vgpr
// ---------------------------------------------------------------------------
__device__ __forceinline__ v16bf frag_from2(const __bf16* p0, const __bf16* p1) {
  union { uint4 u[2]; v16bf v; } c;
  c.u[0] = *reinterpret_cast<const uint4*>(p0);
  c.u[1] = *reinterpret_cast<const uint4*>(p1);
  return c.v;
}
__device__ __forceinline__ v16bf a_frag(const __bf16* row, int hf) {
  return frag_from2(row + hf * 8, row + 16 + hf * 8);
}
__device__ __forceinline__ v16bf b_frag(const __bf16* row, int hf) {
  return frag_from2(row + hf * 8, row + 16 + hf * 8);
}
__device__ __forceinline__ v8f wmma_bf16(v16bf a, v16bf b, v8f c) {
  return __builtin_amdgcn_wmma_f32_16x16x32_bf16(false, a, false, b, (short)0, c,
                                                 false, false);
}

// Async memory->LDS staging (CDNA5 GLOBAL_LOAD_ASYNC_TO_LDS_*, ASYNCcnt).
// LDS operand = low 32 bits of the generic pointer (flat aperture layout).
__device__ __forceinline__ unsigned lds_addr32(const void* p) {
  return (unsigned)(size_t)p;
}
__device__ __forceinline__ void async_copy32B(const void* lds_dst, const void* gsrc) {
  unsigned l = lds_addr32(lds_dst);
  asm volatile("global_load_async_to_lds_b128 %0, %1, off"
               :: "v"(l), "v"(gsrc) : "memory");
  asm volatile("global_load_async_to_lds_b128 %0, %1, off offset:16"
               :: "v"(l), "v"(gsrc) : "memory");
}
__device__ __forceinline__ void wait_async0() {
  asm volatile("s_wait_asynccnt 0x0" ::: "memory");
}

// ---------------------------------------------------------------------------
// fp32 -> bf16 cast
// ---------------------------------------------------------------------------
__global__ void cast_f32_bf16(const float* __restrict__ src, __bf16* __restrict__ dst,
                              size_t n) {
  size_t i = (size_t)blockIdx.x * blockDim.x + threadIdx.x;
  if (i < n) dst[i] = (__bf16)src[i];
}

// ---------------------------------------------------------------------------
// C[M,N] (f32) = A[M,K] (bf16) @ B[K,N] (bf16).  Block tile 128x64, 8 waves,
// each wave a 32x32 patch (2x2 WMMA tiles). K step 32. Double-buffered LDS:
// A tiles staged with async-to-LDS, B tiles through VGPRs (transposed store).
// ---------------------------------------------------------------------------
#define TM 128
#define TN 64
#define TK 32
__global__ __launch_bounds__(256) void gemm_bf16(const __bf16* __restrict__ A,
                                                 const __bf16* __restrict__ Bm,
                                                 float* __restrict__ C,
                                                 int M, int N, int K) {
  __shared__ __bf16 Ash[2][TM][TK + 8];   // row stride 80B (16B aligned)
  __shared__ __bf16 BTsh[2][TN][TK + 8];  // transposed: [n][k]
  const int tid  = threadIdx.x;
  const int lane = tid & 31;
  const int wave = tid >> 5;
  const int hf   = lane >> 4;
  const int ln   = lane & 15;
  const int m0 = blockIdx.y * TM;
  const int n0 = blockIdx.x * TN;
  const int waveM = (wave >> 1) * 32;
  const int waveN = (wave & 1) * 32;

  // staging geometry (per thread: 16 bf16 of A, 8 bf16 of B)
  const int ar = tid >> 1, ac = (tid & 1) * 16;
  const int bk = tid >> 3, bn = (tid & 7) * 8;
  const __bf16* aptr = A + (size_t)(m0 + ar) * K + ac;
  const __bf16* bptr = Bm + (size_t)bk * N + n0 + bn;

  v8f acc[2][2] = {};

  // Prologue: stage tile 0 into buffer 0
  async_copy32B(&Ash[0][ar][ac], aptr);
  uint4 breg = *reinterpret_cast<const uint4*>(bptr);
  {
    const __bf16* e = reinterpret_cast<const __bf16*>(&breg);
#pragma unroll
    for (int j = 0; j < 8; ++j) BTsh[0][bn + j][bk] = e[j];
  }
  wait_async0();
  __syncthreads();

  for (int k0 = 0; k0 < K; k0 += TK) {
    const int buf = (k0 / TK) & 1;
    const int nxt = buf ^ 1;
    const bool more = (k0 + TK) < K;
    if (more) {
      // kick off next tile while we compute on the current one
      async_copy32B(&Ash[nxt][ar][ac], aptr + k0 + TK);
      breg = *reinterpret_cast<const uint4*>(bptr + (size_t)(k0 + TK) * N);
      __builtin_prefetch(bptr + (size_t)(k0 + 2 * TK) * N, 0, 1);
    }
#pragma unroll
    for (int mt = 0; mt < 2; ++mt) {
      v16bf af = a_frag(&Ash[buf][waveM + mt * 16 + ln][0], hf);
#pragma unroll
      for (int nt = 0; nt < 2; ++nt) {
        v16bf bfv = b_frag(&BTsh[buf][waveN + nt * 16 + ln][0], hf);
        acc[mt][nt] = wmma_bf16(af, bfv, acc[mt][nt]);
      }
    }
    if (more) {
      const __bf16* e = reinterpret_cast<const __bf16*>(&breg);
#pragma unroll
      for (int j = 0; j < 8; ++j) BTsh[nxt][bn + j][bk] = e[j];
    }
    wait_async0();
    __syncthreads();
  }

#pragma unroll
  for (int mt = 0; mt < 2; ++mt)
#pragma unroll
    for (int nt = 0; nt < 2; ++nt) {
      int col = n0 + waveN + nt * 16 + ln;
#pragma unroll
      for (int r = 0; r < 8; ++r) {
        int row = m0 + waveM + mt * 16 + hf * 8 + r;
        C[(size_t)row * N + col] = acc[mt][nt][r];
      }
    }
}

// ---------------------------------------------------------------------------
// RoPE + head split: qkv f32 [B,S,3H] -> q/k/v bf16 [B,NH,S,HD].
// One thread per (b,s,h,d<64) rotation pair. Folds 1/sqrt(HD) into q.
// ---------------------------------------------------------------------------
__global__ void rope_split(const float* __restrict__ qkv, const int* __restrict__ pos_ids,
                           __bf16* __restrict__ qo, __bf16* __restrict__ ko,
                           __bf16* __restrict__ vo) {
  size_t idx = (size_t)blockIdx.x * blockDim.x + threadIdx.x;
  size_t total = (size_t)BATCH * SEQ * NH * 64;
  if (idx >= total) return;
  int d = idx & 63;
  int h = (idx >> 6) & (NH - 1);
  int s = (int)((idx >> 11) & (SEQ - 1));
  int b = (int)(idx >> 22);

  const float* row = qkv + (size_t)(b * SEQ + s) * (3 * HIDDEN);
  int col = h * HD + d;
  float q0 = row[col],              q1 = row[col + 64];
  float k0 = row[HIDDEN + col],     k1 = row[HIDDEN + col + 64];
  float v0 = row[2 * HIDDEN + col], v1 = row[2 * HIDDEN + col + 64];

  float p = (float)pos_ids[b * SEQ + s];
  float freq = __powf(10000.0f, -(float)d * (1.0f / 64.0f));
  float sn, cs;
  __sincosf(p * freq, &sn, &cs);
  const float scale = 0.08838834764831845f;  // 1/sqrt(128)

  size_t ob = (((size_t)(b * NH + h)) * SEQ + s) * HD + d;
  qo[ob]      = (__bf16)((q0 * cs - q1 * sn) * scale);
  qo[ob + 64] = (__bf16)((q1 * cs + q0 * sn) * scale);
  ko[ob]      = (__bf16)(k0 * cs - k1 * sn);
  ko[ob + 64] = (__bf16)(k1 * cs + k0 * sn);
  vo[ob]      = (__bf16)v0;
  vo[ob + 64] = (__bf16)v1;
}

// ---------------------------------------------------------------------------
// Causal flash attention. Block = 8 waves x 16 queries = 128 queries per
// (b,head). K staged with async-to-LDS; V transposed via VGPRs. Online
// softmax with cross-lane reductions inside 16-lane groups (C/D striping).
// Output: attn bf16 [B,S,H] ready for the wo GEMM.
// ---------------------------------------------------------------------------
__global__ __launch_bounds__(256) void flash_attn(const __bf16* __restrict__ Q,
                                                  const __bf16* __restrict__ Kt,
                                                  const __bf16* __restrict__ V,
                                                  __bf16* __restrict__ O) {
  __shared__ __bf16 Ksh[32][HD + 8];   // [key][d], row 272B
  __shared__ __bf16 VTsh[HD][32 + 8];  // [d][key], row 80B
  __shared__ __bf16 Psh[8][16][32];    // per-wave P round-trip, row 64B

  const int tid  = threadIdx.x;
  const int lane = tid & 31;
  const int wave = tid >> 5;
  const int hf   = lane >> 4;
  const int ln   = lane & 15;
  const int bh = blockIdx.y;
  const int q0 = blockIdx.x * 128 + wave * 16;

  const __bf16* qbase = Q + (size_t)bh * SEQ * HD;
  const __bf16* kbase = Kt + (size_t)bh * SEQ * HD;
  const __bf16* vbase = V + (size_t)bh * SEQ * HD;

  // Q fragments: 4 chunks of K=32 along head_dim, held for the whole pass
  v16bf qf[4];
  {
    const __bf16* qrow = qbase + (size_t)(q0 + ln) * HD;
#pragma unroll
    for (int c = 0; c < 4; ++c) qf[c] = a_frag(qrow + c * 32, hf);
  }

  v8f oacc[8] = {};
  float mi[8], li[8];
#pragma unroll
  for (int r = 0; r < 8; ++r) { mi[r] = -1e30f; li[r] = 0.0f; }

  const int key = tid >> 3;          // staging geometry: 32 keys x 128 dims
  const int dd  = (tid & 7) * 16;

  const int nkb = (blockIdx.x * 128 + 127) / 32 + 1;
  for (int kb = 0; kb < nkb; ++kb) {
    const int kstart = kb * 32;
    // Stage K via async-to-LDS (pure copy); V via VGPRs (needs transpose)
    async_copy32B(&Ksh[key][dd], kbase + (size_t)(kstart + key) * HD + dd);
    {
      const uint4* gv = reinterpret_cast<const uint4*>(vbase + (size_t)(kstart + key) * HD + dd);
      uint4 d0 = gv[0], d1 = gv[1];
      const __bf16* e0 = reinterpret_cast<const __bf16*>(&d0);
      const __bf16* e1 = reinterpret_cast<const __bf16*>(&d1);
#pragma unroll
      for (int j = 0; j < 8; ++j) {
        VTsh[dd + j][key]     = e0[j];
        VTsh[dd + 8 + j][key] = e1[j];
      }
    }
    wait_async0();
    __syncthreads();

    if (kstart <= q0 + 15) {
      // S = Q K^T : two 16x16 score tiles (keys 0-15, 16-31 of block)
      v8f s0 = {}, s1 = {};
#pragma unroll
      for (int c = 0; c < 4; ++c) {
        v16bf b0 = b_frag(&Ksh[ln][c * 32], hf);
        v16bf b1 = b_frag(&Ksh[16 + ln][c * 32], hf);
        s0 = wmma_bf16(qf[c], b0, s0);
        s1 = wmma_bf16(qf[c], b1, s1);
      }
      // Causal mask + running row max
      float mcur[8];
#pragma unroll
      for (int r = 0; r < 8; ++r) {
        int qg = q0 + hf * 8 + r;
        float a = (kstart + ln)      > qg ? -1e30f : s0[r];
        float b = (kstart + 16 + ln) > qg ? -1e30f : s1[r];
        s0[r] = a; s1[r] = b;
        mcur[r] = fmaxf(a, b);
      }
#pragma unroll
      for (int off = 8; off >= 1; off >>= 1)
#pragma unroll
        for (int r = 0; r < 8; ++r)
          mcur[r] = fmaxf(mcur[r], __shfl_xor(mcur[r], off, 32));
      float alpha[8], rs[8];
#pragma unroll
      for (int r = 0; r < 8; ++r) {
        float mnew = fmaxf(mi[r], mcur[r]);
        alpha[r] = __expf(mi[r] - mnew);
        mi[r] = mnew;
        float p0 = __expf(s0[r] - mnew);
        float p1 = __expf(s1[r] - mnew);
        s0[r] = p0; s1[r] = p1;
        rs[r] = p0 + p1;
      }
#pragma unroll
      for (int off = 8; off >= 1; off >>= 1)
#pragma unroll
        for (int r = 0; r < 8; ++r)
          rs[r] += __shfl_xor(rs[r], off, 32);
#pragma unroll
      for (int r = 0; r < 8; ++r) {
        li[r] = li[r] * alpha[r] + rs[r];
        int row = hf * 8 + r;
        Psh[wave][row][ln]      = (__bf16)s0[r];
        Psh[wave][row][16 + ln] = (__bf16)s1[r];
#pragma unroll
        for (int t = 0; t < 8; ++t) oacc[t][r] *= alpha[r];
      }
      // O += P V  (P re-fragmented through per-wave LDS buffer)
      v16bf pf = a_frag(&Psh[wave][ln][0], hf);
#pragma unroll
      for (int t = 0; t < 8; ++t) {
        v16bf bv = b_frag(&VTsh[t * 16 + ln][0], hf);
        oacc[t] = wmma_bf16(pf, bv, oacc[t]);
      }
    }
    __syncthreads();
  }

  // Epilogue: normalize and scatter to [B, S, H] bf16
  {
    int b = bh >> 5;
    int h = bh & 31;
#pragma unroll
    for (int t = 0; t < 8; ++t)
#pragma unroll
      for (int r = 0; r < 8; ++r) {
        int qrow = q0 + hf * 8 + r;
        size_t orow = (size_t)(b * SEQ + qrow);
        O[orow * HIDDEN + h * HD + t * 16 + ln] = (__bf16)(oacc[t][r] / li[r]);
      }
  }
}

// ---------------------------------------------------------------------------
extern "C" void kernel_launch(void* const* d_in, const int* in_sizes, int n_in,
                              void* d_out, int out_size, void* d_ws, size_t ws_size,
                              hipStream_t stream) {
  const float* hidden = (const float*)d_in[0];
  const float* w_qkv  = (const float*)d_in[1];
  const float* wo     = (const float*)d_in[2];
  const int*   pos    = (const int*)d_in[3];
  float* out = (float*)d_out;

  char* ws = (char*)d_ws;
  size_t off = 0;
  auto alloc = [&](size_t bytes) -> void* {
    void* p = ws + off;
    off += (bytes + 255) & ~(size_t)255;
    return p;
  };
  __bf16* hB    = (__bf16*)alloc((size_t)BS * HIDDEN * 2);
  __bf16* wqkvB = (__bf16*)alloc((size_t)HIDDEN * 3 * HIDDEN * 2);
  __bf16* woB   = (__bf16*)alloc((size_t)HIDDEN * HIDDEN * 2);
  float*  qkvF  = (float*)alloc((size_t)BS * 3 * HIDDEN * 4);
  __bf16* qB    = (__bf16*)alloc((size_t)BS * HIDDEN * 2);
  __bf16* kB    = (__bf16*)alloc((size_t)BS * HIDDEN * 2);
  __bf16* vB    = (__bf16*)alloc((size_t)BS * HIDDEN * 2);
  __bf16* attnB = (__bf16*)alloc((size_t)BS * HIDDEN * 2);

  {
    size_t n = (size_t)BS * HIDDEN;
    cast_f32_bf16<<<(n + 255) / 256, 256, 0, stream>>>(hidden, hB, n);
  }
  {
    size_t n = (size_t)HIDDEN * 3 * HIDDEN;
    cast_f32_bf16<<<(n + 255) / 256, 256, 0, stream>>>(w_qkv, wqkvB, n);
  }
  {
    size_t n = (size_t)HIDDEN * HIDDEN;
    cast_f32_bf16<<<(n + 255) / 256, 256, 0, stream>>>(wo, woB, n);
  }
  // qkv = hidden @ w_qkv
  gemm_bf16<<<dim3(3 * HIDDEN / TN, BS / TM), 256, 0, stream>>>(hB, wqkvB, qkvF,
                                                               BS, 3 * HIDDEN, HIDDEN);
  // RoPE + split into heads
  {
    size_t n = (size_t)BATCH * SEQ * NH * 64;
    rope_split<<<(n + 255) / 256, 256, 0, stream>>>(qkvF, pos, qB, kB, vB);
  }
  // causal flash attention
  flash_attn<<<dim3(SEQ / 128, BATCH * NH), 256, 0, stream>>>(qB, kB, vB, attnB);
  // out = attn @ wo
  gemm_bf16<<<dim3(HIDDEN / TN, BS / TM), 256, 0, stream>>>(attnB, woB, out,
                                                            BS, HIDDEN, HIDDEN);
}
